// FocalCRFLoss_68143951118511
// MI455X (gfx1250) — matrix operand order repaired and checked by reading, not compiled, not run on hardware
//
#include <hip/hip_runtime.h>
#include <hip/hip_bf16.h>

#define Bc 128
#define Tc 2048
#define Kc 64

typedef __attribute__((ext_vector_type(16))) _Float16 v16h;
typedef __attribute__((ext_vector_type(8)))  float    v8f;
typedef __attribute__((__vector_size__(16))) int vi4;
typedef __attribute__((address_space(1))) vi4 gvi4;
typedef __attribute__((address_space(3))) vi4 lvi4;

__device__ __forceinline__ void async_copy16(const void* g, void* l) {
#if __has_builtin(__builtin_amdgcn_global_load_async_to_lds_b128)
    __builtin_amdgcn_global_load_async_to_lds_b128((gvi4*)(void*)g, (lvi4*)l, 0, 0);
#else
    *(float4*)l = *(const float4*)g;   // synchronous fallback
#endif
}

__device__ __forceinline__ void wait_async_le2() {
#if __has_builtin(__builtin_amdgcn_s_wait_asynccnt)
    __builtin_amdgcn_s_wait_asynccnt(2);
#else
    asm volatile("s_wait_asynccnt 0x2" ::: "memory");
#endif
}
__device__ __forceinline__ void wait_async_le0() {
#if __has_builtin(__builtin_amdgcn_s_wait_asynccnt)
    __builtin_amdgcn_s_wait_asynccnt(0);
#else
    asm volatile("s_wait_asynccnt 0x0" ::: "memory");
#endif
}

// ---------------------------------------------------------------------------
// ws layout (floats): [0..127] numerator scores, [128..255] logZ,
//                     [256] focal weight sum, [257] valid count sum
// ---------------------------------------------------------------------------

__global__ void k_init(float* ws) {
    if (threadIdx.x < 2) ws[256 + threadIdx.x] = 0.0f;
}

// -------- CRF forward algorithm (log-partition) with WMMA ------------------
__launch_bounds__(128)
__global__ void k_forward(const float* __restrict__ em,
                          const unsigned char* __restrict__ mask,
                          const float* __restrict__ trans,
                          const float* __restrict__ startT,
                          const float* __restrict__ endT,
                          float* __restrict__ ws) {
    __shared__ float alpha[16][68];                 // 16 batch rows, padded
    __shared__ float rowM[2][16];                   // per-row max (double buffered)
    __shared__ __align__(16) float em_tile[2][16 * 64];  // emissions double buffer
    __shared__ unsigned char m_lds[2][16];

    const int tid  = threadIdx.x;
    const int lane = tid & 31;
    const int wid  = tid >> 5;        // 4 waves: one 16-col N tile each
    const int b0   = blockIdx.x * 16;

    // ---- loop-invariant B fragments: exp(trans), N-tile = wid -------------
    v16h bf[2];
    {
        const int j = wid * 16 + (lane & 15);
        const int g = (lane >> 4) & 1;
#pragma unroll
        for (int tI = 0; tI < 2; ++tI) {
#pragma unroll
            for (int r = 0; r < 8; ++r) {
                const int kk = 2 * r + 16 * g + 32 * tI;
                bf[tI][2 * r]     = (_Float16)__expf(trans[kk * Kc + j]);
                bf[tI][2 * r + 1] = (_Float16)__expf(trans[(kk + 1) * Kc + j]);
            }
        }
    }

    // each thread moves 32 bytes of a 16x64 f32 tile (2 x b128 async)
    auto issue_prefetch = [&](int t, int s) {
        const int rr = tid >> 3;        // row 0..15
        const int cc = (tid & 7) * 8;   // col start
        const float* g = em + ((size_t)(b0 + rr) * Tc + t) * Kc + cc;
        float* l = &em_tile[s][rr * 64 + cc];
        async_copy16(g, l);
        async_copy16(g + 4, l + 4);
    };

    unsigned char mreg = 0;
    if (tid < 16) mreg = mask[(size_t)(b0 + tid) * Tc + 1];
    issue_prefetch(1, 0);

    // ---- alpha0 = start + emissions[:,0,:] --------------------------------
    {
        const int rr = tid >> 3;
        const int cc = (tid & 7) * 8;
        const float* g = em + ((size_t)(b0 + rr) * Tc) * Kc + cc;
#pragma unroll
        for (int i = 0; i < 8; ++i) alpha[rr][cc + i] = startT[cc + i] + g[i];
    }
    __syncthreads();
    if (tid < 16) {
        float m = -1e30f;
#pragma unroll
        for (int j = 0; j < Kc; ++j) m = fmaxf(m, alpha[tid][j]);
        rowM[0][tid] = m;
    }

    // ---- main sequential recurrence over T --------------------------------
    for (int t = 1; t < Tc; ++t) {
        const int s    = (t - 1) & 1;
        const int curM = (t - 1) & 1;
        const int nxtM = t & 1;

        unsigned char mnext = 0;
        if (t + 1 < Tc) {
            issue_prefetch(t + 1, s ^ 1);
            if (tid < 16) mnext = mask[(size_t)(b0 + tid) * Tc + (t + 1)];
        }
        if (tid < 16) m_lds[s][tid] = mreg;
        if (t + 1 < Tc) wait_async_le2(); else wait_async_le0();
        __syncthreads();                                   // B1: stage s ready

        // A fragments: p = exp(alpha - rowM)  (lane owns row M = lane%16)
        const int M  = lane & 15;
        const int g4 = (lane >> 4) & 1;
        const float rm = rowM[curM][M];
        v16h af[2];
#pragma unroll
        for (int tI = 0; tI < 2; ++tI) {
#pragma unroll
            for (int r = 0; r < 8; ++r) {
                const int kk = 2 * r + 16 * g4 + 32 * tI;
                af[tI][2 * r]     = (_Float16)__expf(alpha[M][kk]     - rm);
                af[tI][2 * r + 1] = (_Float16)__expf(alpha[M][kk + 1] - rm);
            }
        }

        // gather epilogue operands (old alpha for mask path, em, rowM, mask)
        float oldA[8], emv[8], rmv[8];
        int mv[8];
        const int col = wid * 16 + (lane & 15);
#pragma unroll
        for (int r = 0; r < 8; ++r) {
            const int Mr = r + 8 * g4;
            oldA[r] = alpha[Mr][col];
            emv[r]  = em_tile[s][Mr * 64 + col];
            rmv[r]  = rowM[curM][Mr];
            mv[r]   = m_lds[s][Mr];
        }

        v8f acc = {};
        acc = __builtin_amdgcn_wmma_f32_16x16x32_f16(false, af[0], false, bf[0],
                                                     (short)0, acc, false, false);
        acc = __builtin_amdgcn_wmma_f32_16x16x32_f16(false, af[1], false, bf[1],
                                                     (short)0, acc, false, false);
        __syncthreads();                                   // B2: alpha reads done

#pragma unroll
        for (int r = 0; r < 8; ++r) {
            const int Mr = r + 8 * g4;
            const float nv = emv[r] + rmv[r] + __logf(acc[r]);
            alpha[Mr][col] = mv[r] ? nv : oldA[r];
        }
        __syncthreads();                                   // B3: alpha written

        if (tid < 16) {
            float m = -1e30f;
#pragma unroll
            for (int j = 0; j < Kc; ++j) m = fmaxf(m, alpha[tid][j]);
            rowM[nxtM][tid] = m;
        }
        mreg = mnext;
    }
    __syncthreads();

    // ---- logZ = logsumexp(alpha + end) ------------------------------------
    if (tid < 16) {
        float m = -1e30f;
#pragma unroll
        for (int j = 0; j < Kc; ++j) m = fmaxf(m, alpha[tid][j] + endT[j]);
        float sum = 0.0f;
#pragma unroll
        for (int j = 0; j < Kc; ++j) sum += __expf(alpha[tid][j] + endT[j] - m);
        ws[128 + b0 + tid] = m + __logf(sum);
    }
}

// -------- gold-path (numerator) score: one thread per sequence -------------
__launch_bounds__(128)
__global__ void k_num(const float* __restrict__ em, const int* __restrict__ lab,
                      const unsigned char* __restrict__ mask,
                      const float* __restrict__ trans,
                      const float* __restrict__ startT,
                      const float* __restrict__ endT, float* ws) {
    const int b = threadIdx.x;
    const int* lb = lab + (size_t)b * Tc;
    const unsigned char* mk = mask + (size_t)b * Tc;
    const float* eb = em + (size_t)b * Tc * Kc;
    int t0 = lb[0];
    int prev = (t0 == -100) ? 0 : t0;
    float score = startT[prev] + eb[prev];
    for (int t = 1; t < Tc; ++t) {
        const int tg = lb[t];
        const int tc2 = (tg == -100) ? 0 : tg;
        const bool v = (tg != -100) && (mk[t] != 0);
        const float step = trans[prev * Kc + tc2] + eb[(size_t)t * Kc + tc2];
        if (v) { score += step; prev = tc2; }
    }
    score += endT[prev];
    ws[b] = score;
}

// -------- focal scale: one wave32 per (b,t) softmax row --------------------
__launch_bounds__(256)
__global__ void k_focal(const float* __restrict__ em, const int* __restrict__ lab,
                        const unsigned char* __restrict__ mask, float* ws) {
    __shared__ float sw[8], sv[8];
    const int lane = threadIdx.x & 31;
    const int wid  = threadIdx.x >> 5;
    const long pos = (long)blockIdx.x * 8 + wid;      // < B*T
    const float* e = em + pos * 64;
    float e0 = e[lane], e1 = e[lane + 32];
    float m = fmaxf(e0, e1);
#pragma unroll
    for (int d = 16; d; d >>= 1) m = fmaxf(m, __shfl_xor(m, d, 32));
    float sum = __expf(e0 - m) + __expf(e1 - m);
#pragma unroll
    for (int d = 16; d; d >>= 1) sum += __shfl_xor(sum, d, 32);
    if (lane == 0) {
        const int lb = lab[pos];
        const int lc = (lb == -100) ? 0 : lb;
        const bool valid = (lb != -100) && (mask[pos] != 0);
        const float p = __expf(e[lc] - m) / sum;
        const float q = 1.0f - p;
        sw[wid] = valid ? q * q : 0.0f;
        sv[wid] = valid ? 1.0f : 0.0f;
    }
    __syncthreads();
    if (threadIdx.x == 0) {
        float a = 0.0f, b = 0.0f;
#pragma unroll
        for (int i = 0; i < 8; ++i) { a += sw[i]; b += sv[i]; }
        atomicAdd(&ws[256], a);
        atomicAdd(&ws[257], b);
    }
}

// -------- final combine ----------------------------------------------------
__launch_bounds__(128)
__global__ void k_final(const float* __restrict__ ws, float* __restrict__ out) {
    __shared__ float red[128];
    const int b = threadIdx.x;
    red[b] = ws[b] - ws[128 + b];          // llh per sequence
    __syncthreads();
    for (int st = 64; st; st >>= 1) {
        if (b < st) red[b] += red[b + st];
        __syncthreads();
    }
    if (b == 0) {
        const float loss  = -red[0] / (float)Bc;
        float scale = ws[256] / fmaxf(ws[257], 1.0f);
        scale = fmaxf(scale, 0.1f);
        out[0] = loss * scale;
    }
}

extern "C" void kernel_launch(void* const* d_in, const int* in_sizes, int n_in,
                              void* d_out, int out_size, void* d_ws, size_t ws_size,
                              hipStream_t stream) {
    const float*         em     = (const float*)d_in[0];
    const int*           lab    = (const int*)d_in[1];
    const unsigned char* mask   = (const unsigned char*)d_in[2];
    const float*         trans  = (const float*)d_in[3];
    const float*         startT = (const float*)d_in[4];
    const float*         endT   = (const float*)d_in[5];
    float* ws  = (float*)d_ws;
    float* out = (float*)d_out;

    k_init<<<1, 32, 0, stream>>>(ws);
    k_focal<<<(Bc * Tc) / 8, 256, 0, stream>>>(em, lab, mask, ws);
    k_num<<<1, 128, 0, stream>>>(em, lab, mask, trans, startT, endT, ws);
    k_forward<<<Bc / 16, 128, 0, stream>>>(em, mask, trans, startT, endT, ws);
    k_final<<<1, 128, 0, stream>>>(ws, out);
}
